// MultiCondEquiUpdate_61211873903539
// MI455X (gfx1250) — matrix operand order repaired and checked by reading, not compiled, not run on hardware
//
#include <hip/hip_runtime.h>

// ---------------------------------------------------------------------------
// MultiCondEquiUpdate fused edge kernel for gfx1250 (MI455X, wave32, WMMA)
// TILE=64 edges/block, 8 waves, 4 M-tiles per wave (B-fragment reuse x4),
// register double-buffered K pipeline (loads for k+1 before WMMAs for k).
// ---------------------------------------------------------------------------

typedef __attribute__((ext_vector_type(16))) _Float16 v16h;
typedef __attribute__((ext_vector_type(8)))  float    v8f;

#define HID     128
#define EDGE_D  64
#define DIST_D  64
#define TIME_D  128
#define IN_CH   384     // 2*HID + EDGE_D + DIST_D
#define HEADS   4
#define TWOH    256
#define TILE    64      // edges per block
#define NTHREADS 256    // 8 wave32

// workspace layout (in _Float16 elements): transposed f16 weights W^T[n][k]
#define WS_TT   0                      // W_time^T : [256][128]
#define WS_TIN  (WS_TT + TWOH*TIME_D)  // W_in^T   : [128][384]
#define WS_T1   (WS_TIN + HID*IN_CH)   // W1^T     : [128][128]
#define WS_END  (WS_T1 + HID*HID)      // = 98304 halves = 192 KiB

// LDS layout (bytes)
#define OFF_A    0        // f16 [64][384] = 49152
                          //  alias: sH2 f32 [64][128] @0 (32768),
                          //         sStat f32 [64][2] @32768 (512),
                          //         sHead f32 [64][4] @33280 (1024)
#define OFF_T    49152    // f16 [64][128] = 16384 ; alias sIH (same size)
#define OFF_TS   65536    // f16 [64][256] = 32768  (shift | scale)
#define OFF_INV  98304    // f32 [64][128] = 32768
#define OFF_IDX  131072   // int sRow[64], sCol[64] = 512
#define OFF_CD   131584   // f32 [64][3] = 768
#define OFF_W2   132352   // f32 [4][128] = 2048 (W2 transposed)
#define SMEM_BYTES 134400

__device__ __forceinline__ float silu_f(float x) { return x / (1.0f + __expf(-x)); }

__device__ __forceinline__ v8f wmma_f16(v16h a, v16h b, v8f c) {
  return __builtin_amdgcn_wmma_f32_16x16x32_f16(false, a, false, b, (short)0, c,
                                                false, false);
}

// A fragment 16x32 f16 from LDS (row-major, stride lda halves).
// ISA layout: lane<16: halves0-7 = K 0..7, halves8-15 = K16..23 ; lane>=16: +8.
__device__ __forceinline__ v16h ld_a(const _Float16* s, int lda, int m0, int k0,
                                     int lane) {
  int r    = m0 + (lane & 15);
  int koff = (lane >> 4) << 3;
  union { v16h h; uint4 q[2]; } u;
  const _Float16* p = s + r * lda + k0 + koff;
  u.q[0] = *(const uint4*)(p);
  u.q[1] = *(const uint4*)(p + 16);
  return u.h;
}

// B fragment 32x16 f16 from global transposed weights W^T[n][k] (stride K).
// lane<16: halves0-7 = K 0..7, halves8-15 = K16..23 of column n ; lane>=16: +8.
__device__ __forceinline__ v16h ld_b(const _Float16* wt, int K, int n0, int k0,
                                     int lane) {
  int n    = n0 + (lane & 15);
  int koff = (lane >> 4) << 3;
  union { v16h h; uint4 q[2]; } u;
  const _Float16* p = wt + n * K + k0 + koff;
  u.q[0] = *(const uint4*)(p);
  u.q[1] = *(const uint4*)(p + 16);
  return u.h;
}

// D tile 16x16 f32 -> LDS. VGPR v holds row m0 + v + 8*(lane>=16), col n0+lane%16.
__device__ __forceinline__ void st_d(float* s, int ldo, int m0, int n0, int lane,
                                     v8f acc) {
  int col = n0 + (lane & 15);
  int rb  = m0 + ((lane >> 4) << 3);
#pragma unroll
  for (int v = 0; v < 8; ++v) s[(rb + v) * ldo + col] = acc[v];
}

__device__ __forceinline__ void st_d_h(_Float16* s, int ldo, int m0, int n0,
                                       int lane, v8f acc) {
  int col = n0 + (lane & 15);
  int rb  = m0 + ((lane >> 4) << 3);
#pragma unroll
  for (int v = 0; v < 8; ++v) s[(rb + v) * ldo + col] = (_Float16)acc[v];
}

// 4-M-tile GEMM, register double-buffered over K:
// fragments for step kb+32 are fetched before the 4 WMMAs of step kb, so each
// s_wait is covered by a full 4-WMMA group and WMMAs issue back-to-back.
template <int K>
__device__ __forceinline__ void gemm4(const _Float16* sAa, const _Float16* wt,
                                      float bias, int n0, int lane, v8f acc[4]) {
#pragma unroll
  for (int m = 0; m < 4; ++m)
#pragma unroll
    for (int v = 0; v < 8; ++v) acc[m][v] = bias;

  v16h b = ld_b(wt, K, n0, 0, lane);
  v16h a[4];
#pragma unroll
  for (int m = 0; m < 4; ++m) a[m] = ld_a(sAa, K, m * 16, 0, lane);

#pragma unroll
  for (int kb = 0; kb < K; kb += 32) {
    v16h bn;
    v16h an[4];
    if (kb + 32 < K) {                 // compile-time resolved
      bn = ld_b(wt, K, n0, kb + 32, lane);
#pragma unroll
      for (int m = 0; m < 4; ++m) an[m] = ld_a(sAa, K, m * 16, kb + 32, lane);
    }
#pragma unroll
    for (int m = 0; m < 4; ++m) acc[m] = wmma_f16(a[m], b, acc[m]);
    if (kb + 32 < K) {
      b = bn;
#pragma unroll
      for (int m = 0; m < 4; ++m) a[m] = an[m];
    }
  }
}

// ---------------------------------------------------------------------------
__global__ __launch_bounds__(NTHREADS) void convert_weights_kernel(
    const float* __restrict__ W_time, const float* __restrict__ W_in,
    const float* __restrict__ W1, _Float16* __restrict__ ws) {
  int i = blockIdx.x * blockDim.x + threadIdx.x;
  if (i < TWOH * TIME_D) {                       // W_time [128][256] -> T[256][128]
    int n = i >> 7, k = i & 127;
    ws[WS_TT + n * TIME_D + k] = (_Float16)W_time[k * TWOH + n];
  } else if (i < TWOH * TIME_D + HID * IN_CH) {  // W_in [384][128] -> T[128][384]
    int j = i - TWOH * TIME_D;
    int n = j / IN_CH, k = j - n * IN_CH;
    ws[WS_TIN + n * IN_CH + k] = (_Float16)W_in[k * HID + n];
  } else if (i < WS_END) {                       // W1 [128][128] -> T[128][128]
    int j = i - (TWOH * TIME_D + HID * IN_CH);
    int n = j >> 7, k = j & 127;
    ws[WS_T1 + n * HID + k] = (_Float16)W1[k * HID + n];
  }
}

__global__ __launch_bounds__(NTHREADS) void init_out_kernel(
    const float* __restrict__ pos, float* __restrict__ out, int n) {
  int i = blockIdx.x * blockDim.x + threadIdx.x;
  if (i < n) out[i] = pos[i];
}

// ---------------------------------------------------------------------------
__global__ __launch_bounds__(NTHREADS) void edge_update_kernel(
    const float* __restrict__ h, const float* __restrict__ pos,
    const float* __restrict__ edge_attr, const float* __restrict__ dist,
    const float* __restrict__ time_emb, const float* __restrict__ adj_extra,
    const float* __restrict__ b_time, const float* __restrict__ b_in,
    const float* __restrict__ b1, const float* __restrict__ W2,
    const float* __restrict__ coors_scale, const int* __restrict__ edge_index,
    const _Float16* __restrict__ wt, float* __restrict__ out, int Ne) {
  extern __shared__ __align__(16) unsigned char smem[];
  _Float16* sA    = (_Float16*)(smem + OFF_A);      // input tile f16 [64][384]
  _Float16* sT    = (_Float16*)(smem + OFF_T);      // silu(time) f16 [64][128]
  _Float16* sTSh  = (_Float16*)(smem + OFF_TS);     // shift|scale f16 [64][256]
  float*    sInv  = (float*)(smem + OFF_INV);       // GEMM1 out f32 [64][128]
  _Float16* sIH   = (_Float16*)(smem + OFF_T);      // modulated f16 (alias sT)
  int*      sRow  = (int*)(smem + OFF_IDX);
  int*      sCol  = (int*)(smem + OFF_IDX + TILE * 4);
  float*    sCD   = (float*)(smem + OFF_CD);
  float*    sW2   = (float*)(smem + OFF_W2);        // [4][128] transposed
  float*    sH2   = (float*)(smem + OFF_A);         // alias (sA dead after GEMM1)
  float*    sStat = (float*)(smem + OFF_A + 32768);
  float*    sHead = (float*)(smem + OFF_A + 33280);

  const int tid  = threadIdx.x;
  const int lane = tid & 31;
  const int w    = tid >> 5;
  const int e0   = blockIdx.x * TILE;
  const float cscale = coors_scale[0];

  // --- edge indices + normalized scaled coordinate diffs ---
  if (tid < TILE) {
    int e = e0 + tid;
    int r = 0, c = 0;
    if (e < Ne) { r = edge_index[e]; c = edge_index[Ne + e]; }
    sRow[tid] = r; sCol[tid] = c;
    float dx = pos[r * 3 + 0] - pos[c * 3 + 0];
    float dy = pos[r * 3 + 1] - pos[c * 3 + 1];
    float dz = pos[r * 3 + 2] - pos[c * 3 + 2];
    float nrm = sqrtf(dx * dx + dy * dy + dz * dz);
    float s = cscale / fmaxf(nrm, 1e-8f);
    sCD[tid * 3 + 0] = dx * s;
    sCD[tid * 3 + 1] = dy * s;
    sCD[tid * 3 + 2] = dz * s;
  }
  // --- stage W2 transposed: sW2[hd][k] ---
  for (int i = tid; i < HEADS * HID; i += NTHREADS) {
    int hd = i & 3, k = i >> 2;
    sW2[hd * HID + k] = W2[k * HEADS + hd];
  }
  __syncthreads();

  // --- stage h_input tile [64][384] as f16, float4-vectorized ---
  // 96 float4 segments per edge: 32 h[row] | 32 h[col] | 16 edge_attr | 16 dist
  for (int i = tid; i < TILE * 96; i += NTHREADS) {
    int el = i / 96, s = i - el * 96;
    int e = e0 + el;
    float4 v = make_float4(0.f, 0.f, 0.f, 0.f);
    if (e < Ne) {
      if (s < 32)       v = *(const float4*)(h + sRow[el] * HID + s * 4);
      else if (s < 64)  v = *(const float4*)(h + sCol[el] * HID + (s - 32) * 4);
      else if (s < 80)  v = *(const float4*)(edge_attr + e * EDGE_D + (s - 64) * 4);
      else              v = *(const float4*)(dist + e * DIST_D + (s - 80) * 4);
    }
    union { _Float16 hh[4]; uint2 u; } pk;
    pk.hh[0] = (_Float16)v.x; pk.hh[1] = (_Float16)v.y;
    pk.hh[2] = (_Float16)v.z; pk.hh[3] = (_Float16)v.w;
    *(uint2*)(sA + el * IN_CH + s * 4) = pk.u;
  }
  // --- stage silu(time_emb) tile [64][128] as f16, float4-vectorized ---
  for (int i = tid; i < TILE * 32; i += NTHREADS) {
    int el = i >> 5, s = i & 31;
    int e = e0 + el;
    float4 v = make_float4(0.f, 0.f, 0.f, 0.f);
    if (e < Ne) v = *(const float4*)(time_emb + e * TIME_D + s * 4);
    union { _Float16 hh[4]; uint2 u; } pk;
    pk.hh[0] = (_Float16)silu_f(v.x); pk.hh[1] = (_Float16)silu_f(v.y);
    pk.hh[2] = (_Float16)silu_f(v.z); pk.hh[3] = (_Float16)silu_f(v.w);
    *(uint2*)(sT + el * TIME_D + s * 4) = pk.u;
  }
  __syncthreads();

  // --- time GEMM: [64][128] x [128][256] -> sTSh (f16), bias b_time ---
  {
    const _Float16* wtT = wt + WS_TT;
#pragma unroll
    for (int rep = 0; rep < 2; ++rep) {
      int n0 = (w + rep * 8) * 16;
      float bias = b_time[n0 + (lane & 15)];
      v8f acc[4];
      gemm4<TIME_D>(sT, wtT, bias, n0, lane, acc);
#pragma unroll
      for (int m = 0; m < 4; ++m) st_d_h(sTSh, TWOH, m * 16, n0, lane, acc[m]);
    }
  }
  // --- GEMM1: [64][384] x [384][128] -> sInv (f32), bias b_in ---
  {
    const _Float16* wi = wt + WS_TIN;
    int n0 = w * 16;
    float bias = b_in[n0 + (lane & 15)];
    v8f acc[4];
    gemm4<IN_CH>(sA, wi, bias, n0, lane, acc);
#pragma unroll
    for (int m = 0; m < 4; ++m) st_d(sInv, HID, m * 16, n0, lane, acc[m]);
  }
  __syncthreads();

  // --- layernorm row stats (population var, eps 1e-6), float4 reads ---
  if (tid < TILE) {
    float s = 0.0f, q = 0.0f;
    for (int c = 0; c < HID; c += 4) {
      float4 x = *(const float4*)(sInv + tid * HID + c);
      s += x.x + x.y + x.z + x.w;
      q += x.x * x.x + x.y * x.y + x.z * x.z + x.w * x.w;
    }
    float mu  = s * (1.0f / HID);
    float var = q * (1.0f / HID) - mu * mu;
    sStat[tid * 2 + 0] = mu;
    sStat[tid * 2 + 1] = rsqrtf(var + 1e-6f);
  }
  __syncthreads();

  // --- normalize + modulate: x*(1+scale)+shift -> f16, vectorized x4 ---
  for (int i = tid; i < TILE * 32; i += NTHREADS) {
    int el = i >> 5, c4 = (i & 31) * 4;
    float mu = sStat[el * 2], rs = sStat[el * 2 + 1];
    float4 x = *(const float4*)(sInv + el * HID + c4);
    union { _Float16 hh[4]; uint2 u; } shp, scp, op;
    shp.u = *(const uint2*)(sTSh + el * TWOH + c4);
    scp.u = *(const uint2*)(sTSh + el * TWOH + HID + c4);
    float xs[4] = {x.x, x.y, x.z, x.w};
#pragma unroll
    for (int j = 0; j < 4; ++j) {
      float xn = (xs[j] - mu) * rs;
      op.hh[j] = (_Float16)(xn * (1.0f + (float)scp.hh[j]) + (float)shp.hh[j]);
    }
    *(uint2*)(sIH + el * HID + c4) = op.u;
  }
  __syncthreads();

  // --- GEMM2: [64][128] x [128][128] -> silu -> sH2 (aliases sA) ---
  {
    const _Float16* w1 = wt + WS_T1;
    int n0 = w * 16;
    float bias = b1[n0 + (lane & 15)];
    v8f acc[4];
    gemm4<HID>(sIH, w1, bias, n0, lane, acc);
#pragma unroll
    for (int m = 0; m < 4; ++m) {
#pragma unroll
      for (int v = 0; v < 8; ++v) acc[m][v] = silu_f(acc[m][v]);
      st_d(sH2, HID, m * 16, n0, lane, acc[m]);
    }
  }
  __syncthreads();

  // --- heads: tanh(h2 @ W2) * adjs ; all 256 threads (64 edges x 4 heads) ---
  {
    int el = tid >> 2, hd = tid & 3;
    int e = e0 + el;
    float s = 0.0f;
    for (int k = 0; k < HID; k += 4) {
      float4 a = *(const float4*)(sH2 + el * HID + k);
      float4 b = *(const float4*)(sW2 + hd * HID + k);
      s += a.x * b.x + a.y * b.y + a.z * b.z + a.w * b.w;
    }
    float th = tanhf(s);
    float adj = 1.0f;
    if (hd > 0 && e < Ne) adj = adj_extra[e * 3 + (hd - 1)];
    sHead[tid] = th * adj;
  }
  __syncthreads();

  // --- mean over heads, scale coord diff, scatter-add into out[row] ---
  if (tid < TILE) {
    int e = e0 + tid;
    if (e < Ne) {
      float inv = 0.25f * (sHead[tid * 4] + sHead[tid * 4 + 1] +
                           sHead[tid * 4 + 2] + sHead[tid * 4 + 3]);
      int r = sRow[tid];
#pragma unroll
      for (int j = 0; j < 3; ++j)
        unsafeAtomicAdd(&out[r * 3 + j], sCD[tid * 3 + j] * inv);
    }
  }
}

// ---------------------------------------------------------------------------
extern "C" void kernel_launch(void* const* d_in, const int* in_sizes, int n_in,
                              void* d_out, int out_size, void* d_ws, size_t ws_size,
                              hipStream_t stream) {
  const float* h           = (const float*)d_in[0];
  const float* pos         = (const float*)d_in[1];
  const float* edge_attr   = (const float*)d_in[2];
  const float* dist        = (const float*)d_in[3];
  const float* time_emb    = (const float*)d_in[4];
  const float* adj_extra   = (const float*)d_in[5];
  const float* W_time      = (const float*)d_in[6];
  const float* b_time      = (const float*)d_in[7];
  const float* W_in        = (const float*)d_in[8];
  const float* b_in        = (const float*)d_in[9];
  const float* W1          = (const float*)d_in[10];
  const float* b1          = (const float*)d_in[11];
  const float* W2          = (const float*)d_in[12];
  const float* coors_scale = (const float*)d_in[13];
  const int*   edge_index  = (const int*)d_in[14];

  int Nn3 = in_sizes[1];       // N * 3
  int Ne  = in_sizes[14] / 2;  // edges

  float*     out = (float*)d_out;
  _Float16*  ws  = (_Float16*)d_ws;

  convert_weights_kernel<<<(WS_END + NTHREADS - 1) / NTHREADS, NTHREADS, 0, stream>>>(
      W_time, W_in, W1, ws);
  init_out_kernel<<<(Nn3 + NTHREADS - 1) / NTHREADS, NTHREADS, 0, stream>>>(
      pos, out, Nn3);
  int blocks = (Ne + TILE - 1) / TILE;
  edge_update_kernel<<<blocks, NTHREADS, SMEM_BYTES, stream>>>(
      h, pos, edge_attr, dist, time_emb, adj_extra, b_time, b_in, b1, W2,
      coors_scale, edge_index, ws, out, Ne);
}
